// SelectiveScan_14791867367992
// MI455X (gfx1250) — matrix operand order repaired and checked by reading, not compiled, not run on hardware
//
#include <hip/hip_runtime.h>

// ---------------- CDNA5 WMMA types ----------------
typedef __attribute__((ext_vector_type(16))) __bf16 v16bf;
typedef __attribute__((ext_vector_type(8)))  float  v8f;

// ---------------- problem constants ----------------
constexpr int NB = 4;                 // batch
constexpr int NS = 8192;              // sequence length
constexpr int ND = 1024;              // channels (D)
constexpr int NR = 64;                // rank (R)
constexpr int CH = 64;                // time chunk per block
constexpr int NCHUNK = NS / CH;       // 128
constexpr int DT = 64;                // d-channels per block
constexpr int NDBLK = ND / DT;        // 16

constexpr float LOG2E = 1.4426950408889634f;

// decay = exp(A * softplus(x)) = 2^(A * log2(1 + 2^(x*log2e)))   (log2e*ln2 == 1)
// Branch-free, uses raw v_exp_f32 / v_log_f32 (log arg >= 1, so no fixup needed).
__device__ __forceinline__ float decay_of(float x, float A) {
  const float t = __builtin_amdgcn_exp2f(x * LOG2E);
  return __builtin_amdgcn_exp2f(A * __builtin_amdgcn_logf(1.f + t));
}

// Pack 16 floats (4x float4) into a bf16 ext-vector (RNE via fp cast).
__device__ __forceinline__ v16bf pack16(float4 a, float4 b, float4 c, float4 d) {
  v16bf v;
  v[0]  = (__bf16)a.x; v[1]  = (__bf16)a.y; v[2]  = (__bf16)a.z; v[3]  = (__bf16)a.w;
  v[4]  = (__bf16)b.x; v[5]  = (__bf16)b.y; v[6]  = (__bf16)b.z; v[7]  = (__bf16)b.w;
  v[8]  = (__bf16)c.x; v[9]  = (__bf16)c.y; v[10] = (__bf16)c.z; v[11] = (__bf16)c.w;
  v[12] = (__bf16)d.x; v[13] = (__bf16)d.y; v[14] = (__bf16)d.z; v[15] = (__bf16)d.w;
  return v;
}

// Compute the CHxDT decay tile for (batch b, time s0.., channels d0..) into LDS.
// Block = 64 threads = 2 waves; wave w owns d-columns [w*32, w*32+32).
// dt = softplus(delta @ W^T + bias);  decay = exp(dt * -exp(A_log))
__device__ __forceinline__ void compute_decay_tile(
    const float* __restrict__ delta, const float* __restrict__ W,
    const float* __restrict__ bias,  const float* __restrict__ A_log,
    int b, int s0, int d0, float (*decayLDS)[DT + 1])
{
  const int tid  = threadIdx.x;
  const int lane = tid & 31;
  const int wave = tid >> 5;
  const int hl   = lane & 15;    // half-lane id (row/col index within 16)
  const int hi   = lane >> 4;    // which half of the wave

  // ---- B operands: W tile, two 16-wide d sub-tiles (tj) x two K-halves ----
  // B 32x16, k order chosen here (not AMD's table order): lane (hl,hi) holds K = hi*16 + [0..15] for column hl.
  v16bf Bop[2][2];
  float biasv[2], Av[2];
#pragma unroll
  for (int j = 0; j < 2; ++j) {
    const int tj   = wave * 2 + j;
    const int dcol = d0 + tj * 16 + hl;
    const float* wrow = W + (size_t)dcol * NR;
    biasv[j] = bias[dcol];
    Av[j]    = -__builtin_amdgcn_exp2f(A_log[dcol] * LOG2E);   // A = -exp(A_log)
#pragma unroll
    for (int kh = 0; kh < 2; ++kh) {
      const float4* p4 = (const float4*)(wrow + kh * 32 + hi * 16);
      Bop[j][kh] = pack16(p4[0], p4[1], p4[2], p4[3]);
    }
  }

  // ---- time tiles ----
#pragma unroll
  for (int ti = 0; ti < 4; ++ti) {
    // A 16x32, same chosen k order (not AMD's table order): lane (hl,hi) holds row hl,
    // K = hi*16 + [0..15] in elems 0..15 (matches B's K order).
    const float* drow = delta + ((size_t)b * NS + s0 + ti * 16 + hl) * NR;
    v16bf Aop[2];
#pragma unroll
    for (int kh = 0; kh < 2; ++kh) {
      const float4* p0 = (const float4*)(drow + kh * 32 + hi * 16);
      const float4* p1 = (const float4*)(drow + kh * 32 + hi * 16 + 8);
      Aop[kh] = pack16(p0[0], p0[1], p1[0], p1[1]);
    }
#pragma unroll
    for (int j = 0; j < 2; ++j) {
      v8f acc;
#pragma unroll
      for (int e = 0; e < 8; ++e) acc[e] = biasv[j];   // fold bias into C
      acc = __builtin_amdgcn_wmma_f32_16x16x32_bf16(
              false, Aop[0], false, Bop[j][0], (short)0, acc, false, false);
      acc = __builtin_amdgcn_wmma_f32_16x16x32_bf16(
              false, Aop[1], false, Bop[j][1], (short)0, acc, false, false);
      // C/D layout: elem e -> row = ti*16 + hi*8 + e, col = tj*16 + hl
      const int col = (wave * 2 + j) * 16 + hl;
#pragma unroll
      for (int e = 0; e < 8; ++e) {
        const int row = ti * 16 + hi * 8 + e;
        decayLDS[row][col] = decay_of(acc[e], Av[j]);
      }
    }
  }
}

// ---- Pass 1: per-chunk local scan + carries + skip connection ----
__global__ __launch_bounds__(64) void ss_chunk_kernel(
    const float* __restrict__ u, const float* __restrict__ delta,
    const float* __restrict__ W, const float* __restrict__ bias,
    const float* __restrict__ A_log, const float* __restrict__ D_skip,
    float* __restrict__ out, float* __restrict__ carryA, float* __restrict__ carryH)
{
  __shared__ float decayLDS[CH][DT + 1];
  __shared__ __attribute__((aligned(16))) float uLDS[CH][DT];

  const int bid  = blockIdx.x;
  const int dblk = bid % NDBLK;
  const int c    = (bid / NDBLK) % NCHUNK;
  const int b    = bid / (NDBLK * NCHUNK);
  const int s0   = c * CH;
  const int d0   = dblk * DT;
  const int tid  = threadIdx.x;

  // stage u tile (coalesced float4: 1 KB per iteration across the block)
  const float* uBase = u + ((size_t)b * NS + s0) * ND + d0;
#pragma unroll
  for (int it = 0; it < 16; ++it) {
    const int idx  = it * 64 + tid;       // 0..1023 float4s
    const int row  = idx >> 4;
    const int col4 = (idx & 15) << 2;
    *(float4*)(&uLDS[row][col4]) =
        *(const float4*)(uBase + (size_t)row * ND + col4);
  }

  compute_decay_tile(delta, W, bias, A_log, b, s0, d0, decayLDS);
  __syncthreads();

  // serial scan over the chunk; thread tid owns channel column tid
  float h = 0.f, prod = 1.f;
  for (int s = 0; s < CH; ++s) {
    const float dec = decayLDS[s][tid];
    h = h * dec + uLDS[s][tid];
    prod *= dec;
    decayLDS[s][tid] = h;                 // reuse LDS as local-h storage
  }
  const size_t ci = (size_t)c * (NB * ND) + (size_t)b * ND + d0 + tid;
  carryA[ci] = prod;
  carryH[ci] = h;
  __syncthreads();

  // out = h_local + u * D_skip  (row-at-a-time: 64 consecutive floats/row)
  const float skipv = D_skip[d0 + tid];
  float* outBase = out + ((size_t)b * NS + s0) * ND + d0;
  for (int s = 0; s < CH; ++s)
    outBase[(size_t)s * ND + tid] = decayLDS[s][tid] + uLDS[s][tid] * skipv;
}

// ---- Pass 2: scan the per-chunk carries -> incoming state per chunk ----
__global__ __launch_bounds__(256) void ss_carry_kernel(
    const float* __restrict__ carryA, const float* __restrict__ carryH,
    float* __restrict__ prefixH)
{
  const int ch = blockIdx.x * blockDim.x + threadIdx.x;   // channel = b*ND + d
  if (ch >= NB * ND) return;
  float state = 0.f;
  for (int c = 0; c < NCHUNK; ++c) {
    const size_t i = (size_t)c * (NB * ND) + ch;
    prefixH[i] = state;                       // state entering chunk c
    state = carryA[i] * state + carryH[i];
  }
}

// ---- Pass 3: out += cumprod(decay) * H_in (recomputes decay tile) ----
__global__ __launch_bounds__(64) void ss_fixup_kernel(
    const float* __restrict__ delta, const float* __restrict__ W,
    const float* __restrict__ bias,  const float* __restrict__ A_log,
    const float* __restrict__ prefixH, float* __restrict__ out)
{
  __shared__ float decayLDS[CH][DT + 1];

  const int bid  = blockIdx.x;
  const int dblk = bid % NDBLK;
  const int c    = (bid / NDBLK) % NCHUNK;
  const int b    = bid / (NDBLK * NCHUNK);
  if (c == 0) return;                        // uniform: incoming state is 0
  const int s0  = c * CH;
  const int d0  = dblk * DT;
  const int tid = threadIdx.x;

  compute_decay_tile(delta, W, bias, A_log, b, s0, d0, decayLDS);
  __syncthreads();

  const float Hin = prefixH[(size_t)c * (NB * ND) + (size_t)b * ND + d0 + tid];
  float p = 1.f;
  float* outBase = out + ((size_t)b * NS + s0) * ND + d0;
  for (int s = 0; s < CH; ++s) {
    p *= decayLDS[s][tid];
    outBase[(size_t)s * ND + tid] += p * Hin;
  }
}

extern "C" void kernel_launch(void* const* d_in, const int* in_sizes, int n_in,
                              void* d_out, int out_size, void* d_ws, size_t ws_size,
                              hipStream_t stream) {
  const float* u      = (const float*)d_in[0];
  const float* delta  = (const float*)d_in[1];
  const float* W      = (const float*)d_in[2];
  const float* bias   = (const float*)d_in[3];
  const float* A_log  = (const float*)d_in[4];
  const float* D_skip = (const float*)d_in[5];
  float* out = (float*)d_out;

  // workspace: carryA | carryH | prefixH, each NCHUNK * NB * ND floats (2 MB)
  float* carryA  = (float*)d_ws;
  float* carryH  = carryA + (size_t)NCHUNK * NB * ND;
  float* prefixH = carryH + (size_t)NCHUNK * NB * ND;

  const int nblk = NB * NCHUNK * NDBLK;   // 8192 blocks, 2 waves each
  ss_chunk_kernel<<<nblk, 64, 0, stream>>>(u, delta, W, bias, A_log, D_skip,
                                           out, carryA, carryH);
  ss_carry_kernel<<<(NB * ND + 255) / 256, 256, 0, stream>>>(carryA, carryH, prefixH);
  ss_fixup_kernel<<<nblk, 64, 0, stream>>>(delta, W, bias, A_log, prefixH, out);
}
